// DecoderWithAttention_33646773796999
// MI455X (gfx1250) — compile-verified, matrix operand order checked by reading, not run on hardware
//
#include <hip/hip_runtime.h>
#include <math.h>

// ---------------------------------------------------------------------------
// Types
// ---------------------------------------------------------------------------
typedef unsigned int       u32;
typedef unsigned short     u16;
typedef __bf16  v16bf __attribute__((ext_vector_type(16)));
typedef float   v8f   __attribute__((ext_vector_type(8)));

__device__ __forceinline__ u16 f2bf(float f) {
    u32 u = __builtin_bit_cast(u32, f);
    u += 0x7FFFu + ((u >> 16) & 1u);   // round-to-nearest-even
    return (u16)(u >> 16);
}
__device__ __forceinline__ float sigm(float x) { return 1.0f / (1.0f + __expf(-x)); }

union FragBF { u32 u[8]; v16bf v; };

// Problem constants
#define BATCH 64
#define PPIX  196
#define ENC   2048
#define DEC   512
#define ATT   512
#define EMB   512
#define VOC   9490
#define VOCP  9536        // VOC padded to multiple of 64
#define KCAT  3072        // EMB + ENC + DEC (packed LSTM input | h)
#define TDEC  20
#define TCAP  21

// ---------------------------------------------------------------------------
// fp32 -> bf16 convert with strided destination (rows x cols -> ldd-strided)
// ---------------------------------------------------------------------------
__global__ __launch_bounds__(256) void k_conv_strided(const float* __restrict__ in,
                                                      u16* __restrict__ out,
                                                      int n, int cols, int ldd) {
    int i = blockIdx.x * 256 + threadIdx.x;
    if (i < n) {
        int r = i / cols, cc = i - r * cols;
        out[(size_t)r * ldd + cc] = f2bf(in[i]);
    }
}

// ---------------------------------------------------------------------------
// bias_cat = a + b (fp32)
// ---------------------------------------------------------------------------
__global__ __launch_bounds__(256) void k_bias_add(const float* __restrict__ a,
                                                  const float* __restrict__ b,
                                                  float* __restrict__ o, int n) {
    int i = blockIdx.x * 256 + threadIdx.x;
    if (i < n) o[i] = a[i] + b[i];
}

// ---------------------------------------------------------------------------
// Weight convert + transpose: W (K x N) fp32 -> Wt (Npad x K) bf16, with
// destination row stride ldd (k-units); zero-padded for n >= N.
// Grid: (K/32, Npad/32). K multiple of 32 required.
// ---------------------------------------------------------------------------
__global__ __launch_bounds__(256) void k_convT_bf16(const float* __restrict__ in,
                                                    u16* __restrict__ out,
                                                    int K, int N, int ldd) {
    __shared__ float tile[32][33];
    const int k0 = blockIdx.x * 32;
    const int n0 = blockIdx.y * 32;
    const int tx = threadIdx.x & 31;
    const int ty = threadIdx.x >> 5;    // 0..7
    #pragma unroll
    for (int r = 0; r < 32; r += 8) {
        int k = k0 + ty + r, n = n0 + tx;
        tile[ty + r][tx] = (n < N) ? in[(size_t)k * N + n] : 0.0f;
    }
    __syncthreads();
    #pragma unroll
    for (int r = 0; r < 32; r += 8) {
        int n = n0 + ty + r, k = k0 + tx;
        out[(size_t)n * ldd + k] = f2bf(tile[tx][ty + r]);
    }
}

// ---------------------------------------------------------------------------
// Transpose-convert encoder_out (B, E, P) fp32 -> enc_bf (B*P, E) bf16
// ---------------------------------------------------------------------------
__global__ __launch_bounds__(256) void k_enc_transpose(const float* __restrict__ in,
                                                       u16* __restrict__ out) {
    __shared__ float tile[32][33];
    const int b  = blockIdx.z;
    const int e0 = blockIdx.x * 32;
    const int p0 = blockIdx.y * 32;
    const int tx = threadIdx.x & 31;
    const int ty = threadIdx.x >> 5;
    const float* src = in + (size_t)b * ENC * PPIX;
    #pragma unroll
    for (int r = 0; r < 32; r += 8) {
        int e = e0 + ty + r, p = p0 + tx;
        tile[ty + r][tx] = (p < PPIX) ? src[(size_t)e * PPIX + p] : 0.0f;
    }
    __syncthreads();
    u16* dst = out + (size_t)b * PPIX * ENC;
    #pragma unroll
    for (int r = 0; r < 32; r += 8) {
        int p = p0 + ty + r, e = e0 + tx;
        if (p < PPIX) dst[(size_t)p * ENC + e] = f2bf(tile[tx][ty + r]);
    }
}

// ---------------------------------------------------------------------------
// Mean over P of encoder_out (B, E, P) -> mean_enc_bf (B, E) bf16
// ---------------------------------------------------------------------------
__global__ __launch_bounds__(256) void k_mean_enc(const float* __restrict__ in,
                                                  u16* __restrict__ mean_bf) {
    int i = blockIdx.x * 256 + threadIdx.x;
    int b = i >> 11, e = i & 2047;
    const float* src = in + ((size_t)b * ENC + e) * PPIX;
    float acc = 0.0f;
    for (int p = 0; p < PPIX; ++p) acc += src[p];
    mean_bf[i] = f2bf(acc * (1.0f / (float)PPIX));
}

// ---------------------------------------------------------------------------
// WMMA GEMM: C[M,N] = A[M,K](bf16, row stride lda)
//                     @ Bt[Npad,K](bf16 row-major = B transposed, row stride ldbt)
//                     + bias[N]
// 64x64 block tile, 4 waves x (1x4) 16x16 WMMA tiles, K-step 32,
// double-buffered LDS staging + global prefetch 2 tiles ahead.
// Requires: M % 64 == 0, K % 32 == 0, grid.y covers Npad/64 (zero-padded Bt).
// ---------------------------------------------------------------------------
__global__ __launch_bounds__(128) void k_gemm_bf16(const u16* __restrict__ A, int lda,
                                                   const u16* __restrict__ Bt, int ldbt,
                                                   float* __restrict__ C, int ldc,
                                                   const float* __restrict__ bias,
                                                   int M, int N, int K) {
    __shared__ u32 sA[2][64][17];   // [buf][m][k/2] bf16 pairs
    __shared__ u32 sB[2][64][17];   // [buf][n][k/2] bf16 pairs
    const int m0    = blockIdx.x * 64;
    const int n0    = blockIdx.y * 64;
    const int t     = threadIdx.x;
    const int wave  = t >> 5;
    const int lane  = t & 31;
    const int mfrag = lane & 15;
    const int khalf = lane >> 4;
    const int row   = t >> 1;            // staging row 0..63
    const int cb    = (t & 1) * 8;       // staging u32 col base

    v8f acc[4];
    #pragma unroll
    for (int j = 0; j < 4; ++j)
        #pragma unroll
        for (int r = 0; r < 8; ++r) acc[j][r] = 0.0f;

    const u32* gA = (const u32*)(A  + (size_t)(m0 + row) * lda  + cb * 2);
    const u32* gB = (const u32*)(Bt + (size_t)(n0 + row) * ldbt + cb * 2);
    const int steps = K >> 5;

    // prologue: stage tile 0
    u32 ra[8], rb[8];
    #pragma unroll
    for (int j = 0; j < 8; ++j) { ra[j] = gA[j]; rb[j] = gB[j]; }
    #pragma unroll
    for (int j = 0; j < 8; ++j) { sA[0][row][cb + j] = ra[j]; sB[0][row][cb + j] = rb[j]; }
    __syncthreads();

    for (int s = 0; s < steps; ++s) {
        const int cur = s & 1;
        const bool more = (s + 1) < steps;
        if (more) {
            const u32* qa = gA + (size_t)(s + 1) * 16;
            const u32* qb = gB + (size_t)(s + 1) * 16;
            #pragma unroll
            for (int j = 0; j < 8; ++j) { ra[j] = qa[j]; rb[j] = qb[j]; }
            if (s + 2 < steps) {
                __builtin_prefetch(gA + (size_t)(s + 2) * 16, 0, 1);
                __builtin_prefetch(gB + (size_t)(s + 2) * 16, 0, 1);
            }
        }

        FragBF a;
        #pragma unroll
        for (int v = 0; v < 8; ++v) {
            int kp = ((v < 4) ? 0 : 8) + khalf * 4 + (v & 3);   // u32 index = k0/2
            a.u[v] = sA[cur][wave * 16 + mfrag][kp];
        }
        #pragma unroll
        for (int jt = 0; jt < 4; ++jt) {
            FragBF b;
            #pragma unroll
            for (int v = 0; v < 8; ++v) {
                int kp = ((v < 4) ? 0 : 8) + khalf * 4 + (v & 3);
                b.u[v] = sB[cur][jt * 16 + mfrag][kp];
            }
            acc[jt] = __builtin_amdgcn_wmma_f32_16x16x32_bf16(
                false, a.v, false, b.v, (short)0, acc[jt], false, false);
        }

        if (more) {
            const int nxt = cur ^ 1;
            #pragma unroll
            for (int j = 0; j < 8; ++j) {
                sA[nxt][row][cb + j] = ra[j];
                sB[nxt][row][cb + j] = rb[j];
            }
            __syncthreads();
        }
    }

    const int mbase = m0 + wave * 16 + khalf * 8;
    const int nl    = lane & 15;
    #pragma unroll
    for (int jt = 0; jt < 4; ++jt) {
        int n = n0 + jt * 16 + nl;
        if (n < N) {
            float bv = bias ? bias[n] : 0.0f;
            #pragma unroll
            for (int r = 0; r < 8; ++r) {
                C[(size_t)(mbase + r) * ldc + n] = acc[jt][r] + bv;
            }
        }
    }
}

// ---------------------------------------------------------------------------
// Attention score + softmax, one block per batch element
// ---------------------------------------------------------------------------
__device__ __forceinline__ float waveMax(float v) {
    #pragma unroll
    for (int o = 16; o; o >>= 1) v = fmaxf(v, __shfl_down(v, o, 32));
    return v;
}
__device__ __forceinline__ float waveSum(float v) {
    #pragma unroll
    for (int o = 16; o; o >>= 1) v += __shfl_down(v, o, 32);
    return v;
}

__global__ __launch_bounds__(256) void k_att_softmax(const float* __restrict__ att1,
                                                     const float* __restrict__ att2,
                                                     const float* __restrict__ wfull,
                                                     const float* __restrict__ bfull,
                                                     float* __restrict__ alpha_ws,
                                                     float* __restrict__ alpha_out, int t) {
    __shared__ float s[PPIX];
    __shared__ float red[8];
    __shared__ float s_mx, s_sum;
    const int b = blockIdx.x;
    const int lane = threadIdx.x & 31, wave = threadIdx.x >> 5;
    const float* a2 = att2 + (size_t)b * ATT;

    for (int p = wave; p < PPIX; p += 8) {
        const float* rowp = att1 + ((size_t)b * PPIX + p) * ATT;
        float acc = 0.0f;
        for (int a = lane; a < ATT; a += 32) {
            float v = rowp[a] + a2[a];
            acc += (v > 0.0f ? v : 0.0f) * wfull[a];
        }
        acc = waveSum(acc);
        if (lane == 0) s[p] = acc + bfull[0];
    }
    __syncthreads();

    float m = -3.0e38f;
    for (int p = threadIdx.x; p < PPIX; p += 256) m = fmaxf(m, s[p]);
    m = waveMax(m);
    if (lane == 0) red[wave] = m;
    __syncthreads();
    if (threadIdx.x == 0) {
        float mm = red[0];
        for (int w = 1; w < 8; ++w) mm = fmaxf(mm, red[w]);
        s_mx = mm;
    }
    __syncthreads();
    float mx = s_mx;
    float ssum = 0.0f;
    for (int p = threadIdx.x; p < PPIX; p += 256) {
        float e = __expf(s[p] - mx);
        s[p] = e;
        ssum += e;
    }
    ssum = waveSum(ssum);
    __syncthreads();
    if (lane == 0) red[wave] = ssum;
    __syncthreads();
    if (threadIdx.x == 0) {
        float acc2 = 0.0f;
        for (int w = 0; w < 8; ++w) acc2 += red[w];
        s_sum = acc2;
    }
    __syncthreads();
    float inv = 1.0f / s_sum;
    for (int p = threadIdx.x; p < PPIX; p += 256) {
        float a = s[p] * inv;
        alpha_ws[(size_t)b * PPIX + p] = a;
        alpha_out[((size_t)b * TDEC + t) * PPIX + p] = a;
    }
}

// ---------------------------------------------------------------------------
// awe[b,e] = sum_p encoder_out[b,e,p] * alpha[b,p]
// ---------------------------------------------------------------------------
__global__ __launch_bounds__(256) void k_awe(const float* __restrict__ enc,
                                             const float* __restrict__ alpha,
                                             float* __restrict__ awe) {
    int b = blockIdx.y;
    int e = blockIdx.x * 256 + threadIdx.x;
    const float* src = enc + ((size_t)b * ENC + e) * PPIX;
    const float* al  = alpha + (size_t)b * PPIX;
    float acc = 0.0f;
    for (int p = 0; p < PPIX; ++p) acc += src[p] * al[p];
    awe[(size_t)b * ENC + e] = acc;
}

// ---------------------------------------------------------------------------
// Build lstm_cat cols [0, EMB+ENC): [emb(x_t) | sigmoid(gate_lin)*awe] (bf16)
// lstm_cat rows have stride KCAT; cols [EMB+ENC, KCAT) hold h (written elsewhere)
// ---------------------------------------------------------------------------
__global__ __launch_bounds__(256) void k_lstm_in(const float* __restrict__ embedding,
                                                 const int* __restrict__ caps,
                                                 const float* __restrict__ gate_lin,
                                                 const float* __restrict__ awe,
                                                 u16* __restrict__ lstm_cat, int t) {
    int i = blockIdx.x * 256 + threadIdx.x;      // 64 * 2560
    int b = i / (EMB + ENC);
    int j = i - b * (EMB + ENC);
    float val;
    if (j < EMB) {
        int tok = caps[b * TCAP + t];
        val = embedding[(size_t)tok * EMB + j];
    } else {
        int e = j - EMB;
        size_t k = (size_t)b * ENC + e;
        val = sigm(gate_lin[k]) * awe[k];
    }
    lstm_cat[(size_t)b * KCAT + j] = f2bf(val);
}

// ---------------------------------------------------------------------------
// LSTM pointwise: gates (B, 4*DEC) order i,f,g,o -> update c, h; emit h_bf
// h_bf destination has row stride hld (packed into lstm_cat tail columns)
// ---------------------------------------------------------------------------
__global__ __launch_bounds__(256) void k_lstm_elem(const float* __restrict__ gates,
                                                   float* __restrict__ h,
                                                   float* __restrict__ c,
                                                   u16* __restrict__ h_bf, int hld) {
    int i = blockIdx.x * 256 + threadIdx.x;      // 64 * 512
    int b = i >> 9, j = i & 511;
    const float* g = gates + (size_t)b * (4 * DEC);
    float gi = sigm(g[j]);
    float gf = sigm(g[DEC + j]);
    float gg = tanhf(g[2 * DEC + j]);
    float go = sigm(g[3 * DEC + j]);
    float cn = gf * c[i] + gi * gg;
    c[i] = cn;
    float hn = go * tanhf(cn);
    h[i] = hn;
    h_bf[(size_t)b * hld + j] = f2bf(hn);
}

// ---------------------------------------------------------------------------
// Host launcher
// ---------------------------------------------------------------------------
static inline size_t alignUp(size_t x) { return (x + 255) & ~(size_t)255; }

extern "C" void kernel_launch(void* const* d_in, const int* in_sizes, int n_in,
                              void* d_out, int out_size, void* d_ws, size_t ws_size,
                              hipStream_t stream) {
    (void)in_sizes; (void)n_in; (void)out_size; (void)ws_size;

    const float* encoder_out = (const float*)d_in[0];
    const int*   caps        = (const int*)  d_in[1];
    const float* W_enc_att   = (const float*)d_in[2];
    const float* b_enc_att   = (const float*)d_in[3];
    const float* W_dec_att   = (const float*)d_in[4];
    const float* b_dec_att   = (const float*)d_in[5];
    const float* W_full_att  = (const float*)d_in[6];
    const float* b_full_att  = (const float*)d_in[7];
    const float* embedding   = (const float*)d_in[8];
    const float* W_ih        = (const float*)d_in[9];
    const float* W_hh        = (const float*)d_in[10];
    const float* b_ih        = (const float*)d_in[11];
    const float* b_hh        = (const float*)d_in[12];
    const float* W_init_h    = (const float*)d_in[13];
    const float* b_init_h    = (const float*)d_in[14];
    const float* W_init_c    = (const float*)d_in[15];
    const float* b_init_c    = (const float*)d_in[16];
    const float* W_f_beta    = (const float*)d_in[17];
    const float* b_f_beta    = (const float*)d_in[18];
    const float* W_fc        = (const float*)d_in[19];
    const float* b_fc        = (const float*)d_in[20];

    float* preds_out = (float*)d_out;                                  // (B, TDEC, V)
    float* alpha_out = (float*)d_out + (size_t)BATCH * TDEC * VOC;     // (B, TDEC, P)

    // ----- workspace layout -----
    char* ws = (char*)d_ws;
    size_t off = 0;
    auto take = [&](size_t bytes) { char* p = ws + off; off = alignUp(off + bytes); return p; };
    u16*   enc_bf    = (u16*)  take((size_t)BATCH * PPIX * ENC * 2);       // (B*P, ENC)
    u16*   wencattT  = (u16*)  take((size_t)ATT * ENC * 2);                // (512, 2048)
    u16*   wdecattT  = (u16*)  take((size_t)ATT * DEC * 2);                // (512, 512)
    u16*   wfbetaT   = (u16*)  take((size_t)ENC * DEC * 2);                // (2048, 512)
    u16*   wcatT     = (u16*)  take((size_t)(4 * DEC) * KCAT * 2);         // (2048, 3072) = [W_ih^T | W_hh^T]
    u16*   wfcT      = (u16*)  take((size_t)VOCP * DEC * 2);               // (9536, 512)
    u16*   winhT     = (u16*)  take((size_t)DEC * ENC * 2);                // (512, 2048)
    u16*   wincT     = (u16*)  take((size_t)DEC * ENC * 2);                // (512, 2048)
    u16*   mean_bf   = (u16*)  take((size_t)BATCH * ENC * 2);
    float* att1      = (float*)take((size_t)BATCH * PPIX * ATT * 4);
    float* att2      = (float*)take((size_t)BATCH * ATT * 4);
    float* h         = (float*)take((size_t)BATCH * DEC * 4);
    float* c         = (float*)take((size_t)BATCH * DEC * 4);
    float* alpha_ws  = (float*)take((size_t)BATCH * PPIX * 4);
    float* awe       = (float*)take((size_t)BATCH * ENC * 4);
    float* gate_lin  = (float*)take((size_t)BATCH * ENC * 4);
    u16*   lstm_cat  = (u16*)  take((size_t)BATCH * KCAT * 2);             // [x_t | awe' | h]
    float* gates     = (float*)take((size_t)BATCH * 4 * DEC * 4);
    float* bias_cat  = (float*)take((size_t)4 * DEC * 4);

    u16* h_bf = lstm_cat + (EMB + ENC);    // h lives in tail columns, row stride KCAT

    // W (K x N) fp32 -> Wt (Npad x K-slice) bf16 at dst row stride ldd
    auto convT = [&](const float* src, u16* dst, int K, int N, int Npad, int ldd) {
        dim3 grid(K / 32, Npad / 32);
        k_convT_bf16<<<grid, 256, 0, stream>>>(src, dst, K, N, ldd);
    };
    auto gemm = [&](const u16* A, int lda, const u16* Bt, int ldbt, float* C, int ldc,
                    const float* bias, int M, int N, int Npad, int K) {
        dim3 grid(M / 64, Npad / 64);
        k_gemm_bf16<<<grid, 128, 0, stream>>>(A, lda, Bt, ldbt, C, ldc, bias, M, N, K);
    };

    // ----- one-time per launch: transpose-convert weights to bf16 -----
    convT(W_enc_att, wencattT,      ENC,       ATT,     ATT,     ENC);
    convT(W_dec_att, wdecattT,      DEC,       ATT,     ATT,     DEC);
    convT(W_f_beta,  wfbetaT,       DEC,       ENC,     ENC,     DEC);
    convT(W_ih,      wcatT,         EMB + ENC, 4 * DEC, 4 * DEC, KCAT);   // cols [0,2560)
    convT(W_hh,      wcatT + (EMB + ENC), DEC, 4 * DEC, 4 * DEC, KCAT);   // cols [2560,3072)
    convT(W_fc,      wfcT,          DEC,       VOC,     VOCP,    DEC);
    convT(W_init_h,  winhT,         ENC,       DEC,     DEC,     ENC);
    convT(W_init_c,  wincT,         ENC,       DEC,     DEC,     ENC);
    k_bias_add<<<(4 * DEC) / 256, 256, 0, stream>>>(b_ih, b_hh, bias_cat, 4 * DEC);

    // enc transpose-convert: (B,E,P) -> (B*P, E) bf16
    {
        dim3 grid(ENC / 32, (PPIX + 31) / 32, BATCH);
        k_enc_transpose<<<grid, 256, 0, stream>>>(encoder_out, enc_bf);
    }
    // mean over P
    k_mean_enc<<<(BATCH * ENC) / 256, 256, 0, stream>>>(encoder_out, mean_bf);

    // h0 / c0
    gemm(mean_bf, ENC, winhT, ENC, h, DEC, b_init_h, BATCH, DEC, DEC, ENC);
    gemm(mean_bf, ENC, wincT, ENC, c, DEC, b_init_c, BATCH, DEC, DEC, ENC);
    k_conv_strided<<<(BATCH * DEC) / 256, 256, 0, stream>>>(h, h_bf, BATCH * DEC, DEC, KCAT);

    // att1 = enc @ W_enc_att + b   (12544 x 512, K = 2048)
    gemm(enc_bf, ENC, wencattT, ENC, att1, ATT, b_enc_att, BATCH * PPIX, ATT, ATT, ENC);

    // ----- recurrent decode -----
    for (int t = 0; t < TDEC; ++t) {
        gemm(h_bf, KCAT, wdecattT, DEC, att2, ATT, b_dec_att, BATCH, ATT, ATT, DEC);
        k_att_softmax<<<BATCH, 256, 0, stream>>>(att1, att2, W_full_att, b_full_att,
                                                 alpha_ws, alpha_out, t);
        {
            dim3 grid(ENC / 256, BATCH);
            k_awe<<<grid, 256, 0, stream>>>(encoder_out, alpha_ws, awe);
        }
        gemm(h_bf, KCAT, wfbetaT, DEC, gate_lin, ENC, b_f_beta, BATCH, ENC, ENC, DEC);
        k_lstm_in<<<(BATCH * (EMB + ENC)) / 256, 256, 0, stream>>>(
            embedding, caps, gate_lin, awe, lstm_cat, t);
        // fused: gates = [lstm_in | h] @ [W_ih ; W_hh] + (b_ih + b_hh)
        gemm(lstm_cat, KCAT, wcatT, KCAT, gates, 4 * DEC, bias_cat,
             BATCH, 4 * DEC, 4 * DEC, KCAT);
        k_lstm_elem<<<(BATCH * DEC) / 256, 256, 0, stream>>>(gates, h, c, h_bf, KCAT);
        gemm(h_bf, KCAT, wfcT, DEC, preds_out + (size_t)t * VOC, TDEC * VOC, b_fc,
             BATCH, VOC, VOCP, DEC);
    }
}
